// GatedAttention_85933705658891
// MI455X (gfx1250) — matrix operand order repaired and checked by reading, not compiled, not run on hardware
//
#include <hip/hip_runtime.h>
#include <hip/hip_bf16.h>
#include <stdint.h>

#define BB   4
#define SS   2048
#define DIN  2048
#define NH   16
#define HD   128
#define NG   4
#define DOUT 2048

typedef __attribute__((ext_vector_type(16))) _Float16 v16h;
typedef __attribute__((ext_vector_type(8)))  _Float16 v8h;
typedef __attribute__((ext_vector_type(8)))  float    v8f;

__device__ __forceinline__ v8f wmma16(v16h a, v16h b, v8f c) {
  return __builtin_amdgcn_wmma_f32_16x16x32_f16(false, a, false, b, (short)0, c, false, false);
}
__device__ __forceinline__ v8f zero8() {
  v8f z;
#pragma unroll
  for (int i = 0; i < 8; ++i) z[i] = 0.f;
  return z;
}
__device__ __forceinline__ v16h frag16(const _Float16* p0, const _Float16* p1) {
  v8h lo = *(const v8h*)p0;
  v8h hi = *(const v8h*)p1;
  v16h r;
#pragma unroll
  for (int i = 0; i < 8; ++i) { r[i] = lo[i]; r[i + 8] = hi[i]; }
  return r;
}
// A-matrix frag 16x32 f16: lane<16 -> K 0-7 & 16-23 of row (lane&15); lane>=16 -> K 8-15 & 24-31
__device__ __forceinline__ v16h load_a(const _Float16* base, int stride, int lane) {
  const _Float16* p = base + (lane & 15) * stride + ((lane >> 4) << 3);
  return frag16(p, p + 16);
}
// B-matrix frag 32x16 f16: source [col][k] contiguous along k; lane<16 -> K 0-7 & 16-23; lane>=16 -> K 8-15 & 24-31
__device__ __forceinline__ v16h load_b(const _Float16* base, int stride, int lane) {
  const _Float16* p = base + (lane & 15) * stride + ((lane >> 4) << 3);
  return frag16(p, p + 16);
}

// Async global->LDS staging (CDNA5 path, ASYNCcnt-tracked). ldsaddr = low 32
// bits of generic shared pointer (== wave-relative LDS offset per aperture rules).
#define ASYNC_LD_B128(LDS32, GADDR, OFFLIT)                                        \
  asm volatile("global_load_async_to_lds_b128 %0, %1, off offset:" OFFLIT          \
               ::"v"(LDS32), "v"(GADDR) : "memory")

// ---------------------------------------------------------------------------
// WMMA GEMM: C[MxN] = A[MxK] (f32 or f16) * B[KxN] (f32)
// MODE 0: store f16; MODE 1: sigmoid -> f16; MODE 2: store f32
// Block tile 128x128, 8 waves (2x4) of 64x32; k-step 64, double-buffered LDS.
// ---------------------------------------------------------------------------
template <int AHALF>
__device__ __forceinline__ void stage_tiles(const void* __restrict__ Av,
                                            const float* __restrict__ Bw,
                                            _Float16* __restrict__ As,
                                            _Float16* __restrict__ Bt,
                                            int mb, int nb, int k0, int K, int N,
                                            int tid) {
  { // A tile 128x64 -> As[row][k], converted to f16
    int row = tid >> 1, kk = (tid & 1) * 32;
    _Float16* d = As + row * 64 + kk;
    if (AHALF) {
      const v8h* s = (const v8h*)((const _Float16*)Av + (size_t)(mb + row) * K + k0 + kk);
#pragma unroll
      for (int q = 0; q < 4; ++q) ((v8h*)d)[q] = s[q];
    } else {
      const float4* s = (const float4*)((const float*)Av + (size_t)(mb + row) * K + k0 + kk);
#pragma unroll
      for (int q = 0; q < 8; ++q) {
        float4 v = s[q];
        d[q * 4 + 0] = (_Float16)v.x; d[q * 4 + 1] = (_Float16)v.y;
        d[q * 4 + 2] = (_Float16)v.z; d[q * 4 + 3] = (_Float16)v.w;
      }
    }
  }
  { // B tile 64x128 -> Bt[n][k] (transposed), converted to f16
    int kk = tid >> 2, n0 = (tid & 3) * 32;
    const float4* s = (const float4*)(Bw + (size_t)(k0 + kk) * N + nb + n0);
#pragma unroll
    for (int q = 0; q < 8; ++q) {
      float4 v = s[q];
      Bt[(n0 + q * 4 + 0) * 64 + kk] = (_Float16)v.x;
      Bt[(n0 + q * 4 + 1) * 64 + kk] = (_Float16)v.y;
      Bt[(n0 + q * 4 + 2) * 64 + kk] = (_Float16)v.z;
      Bt[(n0 + q * 4 + 3) * 64 + kk] = (_Float16)v.w;
    }
  }
}

template <int AHALF, int MODE>
__global__ __launch_bounds__(256) void gemm_kernel(const void* __restrict__ Av,
                                                   const float* __restrict__ Bw,
                                                   void* __restrict__ Cv,
                                                   int M, int N, int K) {
  __shared__ __align__(16) _Float16 As[2][128 * 64];
  __shared__ __align__(16) _Float16 Bt[2][128 * 64];
  const int tid = threadIdx.x, lane = tid & 31, w = tid >> 5;
  const int wm = (w >> 2) * 64, wn = (w & 3) * 32;
  const int mb = blockIdx.y * 128, nb = blockIdx.x * 128;

  v8f acc[4][2];
#pragma unroll
  for (int i = 0; i < 4; ++i)
#pragma unroll
    for (int j = 0; j < 2; ++j) acc[i][j] = zero8();

  const int nsteps = K / 64;
  stage_tiles<AHALF>(Av, Bw, As[0], Bt[0], mb, nb, 0, K, N, tid);

  for (int st = 0; st < nsteps; ++st) {
    __syncthreads(); // all stores for buffer (st&1) are complete everywhere
    const int cur = st & 1;
    if (st + 1 < nsteps)
      stage_tiles<AHALF>(Av, Bw, As[cur ^ 1], Bt[cur ^ 1], mb, nb, (st + 1) * 64, K, N, tid);
    if (st + 2 < nsteps) { // keep L2->WGP$ warm two tiles ahead
      int pk = (st + 2) * 64;
      if (AHALF)
        __builtin_prefetch((const _Float16*)Av + (size_t)(mb + (tid >> 1)) * K + pk, 0, 0);
      else
        __builtin_prefetch((const float*)Av + (size_t)(mb + (tid >> 1)) * K + pk, 0, 0);
      __builtin_prefetch(Bw + (size_t)(pk + (tid >> 2)) * N + nb, 0, 0);
    }
    const _Float16* Ab = As[cur];
    const _Float16* Bb = Bt[cur];
#pragma unroll
    for (int c = 0; c < 2; ++c) {
      v16h bf[2];
#pragma unroll
      for (int j = 0; j < 2; ++j) bf[j] = load_b(Bb + (wn + j * 16) * 64 + c * 32, 64, lane);
#pragma unroll
      for (int i = 0; i < 4; ++i) {
        v16h af = load_a(Ab + (wm + i * 16) * 64 + c * 32, 64, lane);
#pragma unroll
        for (int j = 0; j < 2; ++j) acc[i][j] = wmma16(af, bf[j], acc[i][j]);
      }
    }
  }
  // epilogue: D-layout element (M = r + 8*(lane>=16), N = lane&15)
#pragma unroll
  for (int i = 0; i < 4; ++i)
#pragma unroll
    for (int j = 0; j < 2; ++j)
#pragma unroll
      for (int r = 0; r < 8; ++r) {
        int row = mb + wm + i * 16 + r + ((lane >> 4) << 3);
        int col = nb + wn + j * 16 + (lane & 15);
        float v = acc[i][j][r];
        if (MODE == 1) v = 1.f / (1.f + __expf(-v));
        if (MODE == 2)
          ((float*)Cv)[(size_t)row * N + col] = v;
        else
          ((_Float16*)Cv)[(size_t)row * N + col] = (_Float16)v;
      }
}

// ---------------------------------------------------------------------------
// Zero-centered RMSNorm + RoPE, in-place on f16 [rows][128]; one wave per row.
// ---------------------------------------------------------------------------
__global__ __launch_bounds__(256) void rmsrope_kernel(_Float16* __restrict__ X,
                                                      const float* __restrict__ cosp,
                                                      const float* __restrict__ sinp,
                                                      const float* __restrict__ scale,
                                                      int nh) {
  const int lane = threadIdx.x & 31, w = threadIdx.x >> 5;
  const int row = blockIdx.x * 8 + w;
  const int s = (row / nh) % SS;
  _Float16* p = X + (size_t)row * HD + lane * 4;
  const int d0 = lane * 4;
  float x[4];
#pragma unroll
  for (int j = 0; j < 4; ++j) x[j] = (float)p[j];
  float ss = x[0] * x[0] + x[1] * x[1] + x[2] * x[2] + x[3] * x[3];
#pragma unroll
  for (int d = 1; d < 32; d <<= 1) ss += __shfl_xor(ss, d);
  const float f = rsqrtf(ss * (1.0f / HD) + 1e-6f);
  float y[4];
#pragma unroll
  for (int j = 0; j < 4; ++j) y[j] = x[j] * f * (1.0f + scale[d0 + j]);
  float part[4];
#pragma unroll
  for (int j = 0; j < 4; ++j) part[j] = __shfl_xor(y[j], 16); // pairs d <-> d+64
  const float sgn = (lane < 16) ? -1.f : 1.f;
#pragma unroll
  for (int j = 0; j < 4; ++j) {
    float cs = cosp[(size_t)s * HD + d0 + j];
    float sn = sinp[(size_t)s * HD + d0 + j];
    p[j] = (_Float16)(y[j] * cs + sgn * part[j] * sn);
  }
}

// ---------------------------------------------------------------------------
// Flash attention (causal + attn_mask), GQA 4:1, fused gate multiply.
// Block: 64 q-rows, 4 waves; wave owns 16 rows. KV tile 64.
// K tile staged with async global->LDS (ASYNCcnt); V staged transposed via VGPRs.
// ---------------------------------------------------------------------------
__global__ __launch_bounds__(128) void attn_kernel(const _Float16* __restrict__ Qp,
                                                   const _Float16* __restrict__ Kp,
                                                   const _Float16* __restrict__ Vp,
                                                   const _Float16* __restrict__ Gp,
                                                   const unsigned char* __restrict__ amask,
                                                   _Float16* __restrict__ ctx) {
  __shared__ __align__(16) _Float16 Klds[64 * HD];  // [kv][hd]
  __shared__ __align__(16) _Float16 Vt[HD * 64];    // [hd][kv]
  __shared__ __align__(16) _Float16 Plds[4 * 16 * 64];
  const int tid = threadIdx.x, lane = tid & 31, w = tid >> 5;
  const int qt = blockIdx.x, h = blockIdx.y, bb = blockIdx.z;
  const int g = h >> 2;
  const int q0 = qt * 64;
  const float SCALE = 0.08838834764831845f; // 1/sqrt(128)

  // Q fragments for this wave's 16 rows (A-layout), 4 chunks of K=32
  v16h aq[4];
  {
    const _Float16* qb = Qp + (((size_t)(bb * SS + q0 + w * 16)) * NH + h) * HD;
#pragma unroll
    for (int c = 0; c < 4; ++c) aq[c] = load_a(qb + c * 32, NH * HD, lane);
  }

  float mi[8], li[8];
  v8f oacc[8];
#pragma unroll
  for (int r = 0; r < 8; ++r) { mi[r] = -1e30f; li[r] = 0.f; }
#pragma unroll
  for (int n = 0; n < 8; ++n) oacc[n] = zero8();

  const int ntiles = qt + 1;
  for (int t = 0; t < ntiles; ++t) {
    const int kvb = t * 64;
    __syncthreads(); // previous tile fully consumed by all waves
    { // cooperative K/V tile staging (128 threads; lane handles 128B)
      int row = tid >> 1, h0 = (tid & 1) * 64;
      const _Float16* ksrc = Kp + (((size_t)(bb * SS + kvb + row)) * NG + g) * HD + h0;
      unsigned ldsK = (unsigned)(uintptr_t)(Klds + row * HD + h0);
      unsigned long long ga = (unsigned long long)(uintptr_t)ksrc;
      ASYNC_LD_B128(ldsK, ga, "0");
      ASYNC_LD_B128(ldsK, ga, "16");
      ASYNC_LD_B128(ldsK, ga, "32");
      ASYNC_LD_B128(ldsK, ga, "48");
      ASYNC_LD_B128(ldsK, ga, "64");
      ASYNC_LD_B128(ldsK, ga, "80");
      ASYNC_LD_B128(ldsK, ga, "96");
      ASYNC_LD_B128(ldsK, ga, "112");
      const _Float16* vs = Vp + (((size_t)(bb * SS + kvb + row)) * NG + g) * HD + h0;
#pragma unroll
      for (int i = 0; i < 64; ++i) Vt[(h0 + i) * 64 + row] = vs[i]; // transpose
      asm volatile("s_wait_asynccnt 0" ::: "memory"); // barrier only waits ds/load cnts
    }
    __syncthreads();

    // scores: 4 subtiles of 16 kv cols
    float scf[4][8];
#pragma unroll
    for (int j = 0; j < 4; ++j) {
      v8f s = zero8();
#pragma unroll
      for (int c = 0; c < 4; ++c) {
        v16h bf = load_b(Klds + (j * 16) * HD + c * 32, HD, lane);
        s = wmma16(aq[c], bf, s);
      }
      const int kvcol = kvb + j * 16 + (lane & 15);
      const unsigned char am = amask[(size_t)bb * SS + kvcol];
#pragma unroll
      for (int r = 0; r < 8; ++r) {
        int qrow = q0 + w * 16 + r + ((lane >> 4) << 3);
        bool msk = (kvcol > qrow) || (am == 0);
        scf[j][r] = msk ? -1e9f : s[r] * SCALE;
      }
    }
    // online softmax (row reductions stay inside 16-lane groups)
    float alpha[8], ps[8];
#pragma unroll
    for (int r = 0; r < 8; ++r) {
      float mx = fmaxf(fmaxf(scf[0][r], scf[1][r]), fmaxf(scf[2][r], scf[3][r]));
#pragma unroll
      for (int d = 1; d < 16; d <<= 1) mx = fmaxf(mx, __shfl_xor(mx, d));
      float mn = fmaxf(mi[r], mx);
      alpha[r] = __expf(mi[r] - mn);
      mi[r] = mn;
      ps[r] = 0.f;
    }
#pragma unroll
    for (int j = 0; j < 4; ++j)
#pragma unroll
      for (int r = 0; r < 8; ++r) {
        float pv = __expf(scf[j][r] - mi[r]);
        scf[j][r] = pv;
        ps[r] += pv;
      }
#pragma unroll
    for (int r = 0; r < 8; ++r) {
#pragma unroll
      for (int d = 1; d < 16; d <<= 1) ps[r] += __shfl_xor(ps[r], d);
      li[r] = li[r] * alpha[r] + ps[r];
    }
#pragma unroll
    for (int n = 0; n < 8; ++n)
#pragma unroll
      for (int r = 0; r < 8; ++r) oacc[n][r] *= alpha[r];

    // P: D-layout -> A-layout via wave-private LDS
    _Float16* pw = Plds + w * 16 * 64;
#pragma unroll
    for (int j = 0; j < 4; ++j)
#pragma unroll
      for (int r = 0; r < 8; ++r)
        pw[(r + ((lane >> 4) << 3)) * 64 + j * 16 + (lane & 15)] = (_Float16)scf[j][r];
    asm volatile("s_wait_dscnt 0" ::: "memory");

    // O += P @ V
#pragma unroll
    for (int c2 = 0; c2 < 2; ++c2) {
      v16h ap = load_a(pw + c2 * 32, 64, lane);
#pragma unroll
      for (int n = 0; n < 8; ++n) {
        v16h bv = load_b(Vt + (n * 16) * 64 + c2 * 32, 64, lane);
        oacc[n] = wmma16(ap, bv, oacc[n]);
      }
    }
  }

  // epilogue: normalize, multiply gate, store ctx (f16)
  float inv[8];
#pragma unroll
  for (int r = 0; r < 8; ++r) inv[r] = 1.f / li[r];
#pragma unroll
  for (int n = 0; n < 8; ++n)
#pragma unroll
    for (int r = 0; r < 8; ++r) {
      int qrow = q0 + w * 16 + r + ((lane >> 4) << 3);
      int col = n * 16 + (lane & 15);
      size_t idx = (((size_t)bb * SS + qrow) * NH + h) * HD + col;
      float gv = (float)Gp[idx];
      ctx[idx] = (_Float16)(oacc[n][r] * inv[r] * gv);
    }
}

// ---------------------------------------------------------------------------
extern "C" void kernel_launch(void* const* d_in, const int* in_sizes, int n_in,
                              void* d_out, int out_size, void* d_ws, size_t ws_size,
                              hipStream_t stream) {
  (void)in_sizes; (void)n_in; (void)out_size; (void)ws_size;
  const float* x = (const float*)d_in[0];
  const unsigned char* amask = (const unsigned char*)d_in[2];
  const float* cosp = (const float*)d_in[3];
  const float* sinp = (const float*)d_in[4];
  const float* Wq = (const float*)d_in[5];
  const float* Wk = (const float*)d_in[6];
  const float* Wv = (const float*)d_in[7];
  const float* Wg = (const float*)d_in[8];
  const float* Wo = (const float*)d_in[9];
  const float* qs = (const float*)d_in[10];
  const float* ks = (const float*)d_in[11];

  const size_t M = (size_t)BB * SS; // 8192 tokens
  _Float16* Q   = (_Float16*)d_ws;
  _Float16* Kx  = Q  + M * DOUT;       // 8192x512 (K proj)
  _Float16* Vx  = Kx + M * (NG * HD);
  _Float16* Gt  = Vx + M * (NG * HD);  // gate, 8192x2048
  _Float16* Ctx = Gt + M * DOUT;       // gated context

  dim3 blk(256);
  gemm_kernel<0, 0><<<dim3(16, 64), blk, 0, stream>>>(x, Wq, Q, 8192, 2048, 2048);
  gemm_kernel<0, 0><<<dim3(4, 64), blk, 0, stream>>>(x, Wk, Kx, 8192, 512, 2048);
  gemm_kernel<0, 0><<<dim3(4, 64), blk, 0, stream>>>(x, Wv, Vx, 8192, 512, 2048);
  gemm_kernel<0, 1><<<dim3(16, 64), blk, 0, stream>>>(x, Wg, Gt, 8192, 2048, 2048);

  rmsrope_kernel<<<(BB * SS * NH) / 8, 256, 0, stream>>>(Q, cosp, sinp, qs, NH);
  rmsrope_kernel<<<(BB * SS * NG) / 8, 256, 0, stream>>>(Kx, cosp, sinp, ks, NG);

  attn_kernel<<<dim3(SS / 64, NH, BB), dim3(128), 0, stream>>>(Q, Kx, Vx, Gt, amask, Ctx);

  gemm_kernel<1, 2><<<dim3(16, 64), blk, 0, stream>>>(Ctx, Wo, d_out, 8192, 2048, 2048);
}
